// GraphAttentionLayer_40802189312767
// MI455X (gfx1250) — matrix operand order repaired and checked by reading, not compiled, not run on hardware
//
#include <hip/hip_runtime.h>

typedef __attribute__((ext_vector_type(16))) _Float16 v16h;
typedef __attribute__((ext_vector_type(8)))  _Float16 v8h;
typedef __attribute__((ext_vector_type(8)))  float    v8f;
typedef __attribute__((ext_vector_type(4)))  int      v4i;
typedef __attribute__((ext_vector_type(4)))  float    v4f;

#define GAT_B   8
#define GAT_N   4096
#define GAT_FI  128
#define GAT_FO  64
#define NEG_BIG (-1.0e12f)
#define LRELU_A 0.2f

// ---------------------------------------------------------------------------
// Kernel A: h = inp @ W (per-thread row), emit h^T as f16 [b][o][n], plus
// s1 = h@a1, s2 = h@a2.  W/a indices are wave-uniform -> scalar (K$) loads.
// ---------------------------------------------------------------------------
__global__ __launch_bounds__(256)
void gat_proj(const float* __restrict__ inp, const float* __restrict__ W,
              const float* __restrict__ a, _Float16* __restrict__ hT,
              float* __restrict__ s1, float* __restrict__ s2)
{
    const int tid = blockIdx.x * 256 + threadIdx.x;      // 0 .. B*N-1
    const int b = tid >> 12;
    const int n = tid & (GAT_N - 1);

    const float4* rowv = reinterpret_cast<const float4*>(inp + (size_t)tid * GAT_FI);

    float acc[GAT_FO];
#pragma unroll
    for (int o = 0; o < GAT_FO; ++o) acc[o] = 0.0f;

    for (int f4 = 0; f4 < GAT_FI / 4; ++f4) {
        const float4 x = rowv[f4];
        const float* wr = W + f4 * 4 * GAT_FO;           // uniform -> s_load
#pragma unroll
        for (int o = 0; o < GAT_FO; ++o) {
            float t = fmaf(x.x, wr[o], acc[o]);
            t = fmaf(x.y, wr[GAT_FO + o], t);
            t = fmaf(x.z, wr[2 * GAT_FO + o], t);
            acc[o] = fmaf(x.w, wr[3 * GAT_FO + o], t);
        }
    }

    float s1a = 0.0f, s2a = 0.0f;
#pragma unroll
    for (int o = 0; o < GAT_FO; ++o) {
        s1a = fmaf(acc[o], a[o], s1a);
        s2a = fmaf(acc[o], a[GAT_FO + o], s2a);
        hT[((size_t)(b * GAT_FO + o) << 12) + n] = (_Float16)acc[o];
    }
    s1[tid] = s1a;
    s2[tid] = s2a;
}

// ---------------------------------------------------------------------------
// Kernel B: fused mask + leakyrelu + online softmax + att@h (WMMA f16->f32)
// + ELU.  One wave per 16-row i-tile; j streamed in chunks of 32 (= WMMA K).
// adj is a read-once 512MB stream -> non-temporal loads + WGP-scope prefetch
// so it does not evict the hot h^T (4MB, reused by every tile) from L2.
// ---------------------------------------------------------------------------
__device__ __forceinline__ float gat_edge(int av, float s2v, float s1r) {
    float e = s1r + s2v;
    e = (e > 0.0f) ? e : LRELU_A * e;
    return (av > 0) ? e : NEG_BIG;
}

__global__ __launch_bounds__(128)
void gat_attn(const int* __restrict__ adj, const _Float16* __restrict__ hT,
              const float* __restrict__ s1g, const float* __restrict__ s2g,
              float* __restrict__ out)
{
    __shared__ float s2s[GAT_N];                         // 16 KB

    const int lane = threadIdx.x & 31;
    const int wave = threadIdx.x >> 5;
    const int b    = blockIdx.y;
    const int i0   = blockIdx.x * 64 + wave * 16;

    {   // stage s2[b,:] into LDS once per block
        const float4* src = reinterpret_cast<const float4*>(s2g + (size_t)b * GAT_N);
        float4* dst = reinterpret_cast<float4*>(s2s);
        for (int k = threadIdx.x; k < GAT_N / 4; k += 128) dst[k] = src[k];
    }
    __syncthreads();

    const int mrow = lane & 15;   // row M served by this lane (A operand); column N for B/C
    const int half = lane >> 4;   // 0: K {0..7,16..23}; 1: K {8..15,24..31}
    const float s1r = s1g[(size_t)b * GAT_N + i0 + mrow];
    const int* adjrow = adj + ((size_t)b * GAT_N + i0 + mrow) * (size_t)GAT_N;

    // h^T row base pointers for the 4 column tiles (n = c*16 + mrow)
    const _Float16* hb0 = hT + ((size_t)(b * GAT_FO +  0 + mrow) << 12);
    const _Float16* hb1 = hT + ((size_t)(b * GAT_FO + 16 + mrow) << 12);
    const _Float16* hb2 = hT + ((size_t)(b * GAT_FO + 32 + mrow) << 12);
    const _Float16* hb3 = hT + ((size_t)(b * GAT_FO + 48 + mrow) << 12);

    float m = -1.0e30f, l = 0.0f;
    v8f acc0 = {}, acc1 = {}, acc2 = {}, acc3 = {};

    for (int jc = 0; jc < GAT_N; jc += 32) {
        if (jc + 512 < GAT_N) __builtin_prefetch(adjrow + jc + 512, 0, 3);

        const int ja = jc + half * 8;    // K run for A elements 0..7
        const int jb = ja + 16;          // K run for A elements 8..15

        // streaming (non-temporal) adjacency loads
        const v4i av0 = __builtin_nontemporal_load(reinterpret_cast<const v4i*>(adjrow + ja));
        const v4i av1 = __builtin_nontemporal_load(reinterpret_cast<const v4i*>(adjrow + ja + 4));
        const v4i av2 = __builtin_nontemporal_load(reinterpret_cast<const v4i*>(adjrow + jb));
        const v4i av3 = __builtin_nontemporal_load(reinterpret_cast<const v4i*>(adjrow + jb + 4));
        const v4f sv0 = *reinterpret_cast<const v4f*>(&s2s[ja]);
        const v4f sv1 = *reinterpret_cast<const v4f*>(&s2s[ja + 4]);
        const v4f sv2 = *reinterpret_cast<const v4f*>(&s2s[jb]);
        const v4f sv3 = *reinterpret_cast<const v4f*>(&s2s[jb + 4]);

        float v[16];
        v[0]  = gat_edge(av0.x, sv0.x, s1r);
        v[1]  = gat_edge(av0.y, sv0.y, s1r);
        v[2]  = gat_edge(av0.z, sv0.z, s1r);
        v[3]  = gat_edge(av0.w, sv0.w, s1r);
        v[4]  = gat_edge(av1.x, sv1.x, s1r);
        v[5]  = gat_edge(av1.y, sv1.y, s1r);
        v[6]  = gat_edge(av1.z, sv1.z, s1r);
        v[7]  = gat_edge(av1.w, sv1.w, s1r);
        v[8]  = gat_edge(av2.x, sv2.x, s1r);
        v[9]  = gat_edge(av2.y, sv2.y, s1r);
        v[10] = gat_edge(av2.z, sv2.z, s1r);
        v[11] = gat_edge(av2.w, sv2.w, s1r);
        v[12] = gat_edge(av3.x, sv3.x, s1r);
        v[13] = gat_edge(av3.y, sv3.y, s1r);
        v[14] = gat_edge(av3.z, sv3.z, s1r);
        v[15] = gat_edge(av3.w, sv3.w, s1r);

        // chunk row-max (row split across lanes L and L+16)
        float cm = v[0];
#pragma unroll
        for (int u = 1; u < 16; ++u) cm = fmaxf(cm, v[u]);
        cm = fmaxf(cm, __shfl_xor(cm, 16, 32));

        const float mnew  = fmaxf(m, cm);
        const float scale = __expf(m - mnew);

        float p[16];
        float ls = 0.0f;
#pragma unroll
        for (int u = 0; u < 16; ++u) { p[u] = __expf(v[u] - mnew); ls += p[u]; }
        ls += __shfl_xor(ls, 16, 32);
        l = l * scale + ls;
        m = mnew;

        // rescale accumulators by per-row scale (C layout: VGPR r <-> M=r / r+8)
        if (__any(scale < 1.0f)) {
            const int srcb = (lane < 16) ? 0 : 8;
#pragma unroll
            for (int r = 0; r < 8; ++r) {
                const float sr = __shfl(scale, srcb + r, 32);
                acc0[r] *= sr; acc1[r] *= sr; acc2[r] *= sr; acc3[r] *= sr;
            }
        }

        // A operand: 16 probabilities packed per the 16-bit A 16x32 layout
        v16h A;
#pragma unroll
        for (int u = 0; u < 16; ++u) A[u] = (_Float16)p[u];

        // B operands: lanes 0-15 carry K=jc+{0..7,16..23}, lanes 16-31 K=jc+{8..15,24..31}
        const int kb = jc + half * 8;
        v8h t0, t1;
        t0 = *reinterpret_cast<const v8h*>(hb0 + kb);
        t1 = *reinterpret_cast<const v8h*>(hb0 + kb + 16);
        const v16h B0 = __builtin_shufflevector(t0, t1, 0,1,2,3,4,5,6,7,8,9,10,11,12,13,14,15);
        t0 = *reinterpret_cast<const v8h*>(hb1 + kb);
        t1 = *reinterpret_cast<const v8h*>(hb1 + kb + 16);
        const v16h B1 = __builtin_shufflevector(t0, t1, 0,1,2,3,4,5,6,7,8,9,10,11,12,13,14,15);
        t0 = *reinterpret_cast<const v8h*>(hb2 + kb);
        t1 = *reinterpret_cast<const v8h*>(hb2 + kb + 16);
        const v16h B2 = __builtin_shufflevector(t0, t1, 0,1,2,3,4,5,6,7,8,9,10,11,12,13,14,15);
        t0 = *reinterpret_cast<const v8h*>(hb3 + kb);
        t1 = *reinterpret_cast<const v8h*>(hb3 + kb + 16);
        const v16h B3 = __builtin_shufflevector(t0, t1, 0,1,2,3,4,5,6,7,8,9,10,11,12,13,14,15);

        acc0 = __builtin_amdgcn_wmma_f32_16x16x32_f16(false, A, false, B0, (short)0, acc0, false, false);
        acc1 = __builtin_amdgcn_wmma_f32_16x16x32_f16(false, A, false, B1, (short)0, acc1, false, false);
        acc2 = __builtin_amdgcn_wmma_f32_16x16x32_f16(false, A, false, B2, (short)0, acc2, false, false);
        acc3 = __builtin_amdgcn_wmma_f32_16x16x32_f16(false, A, false, B3, (short)0, acc3, false, false);
    }

    // finalize: divide by row sum, apply ELU, store (write-once -> non-temporal)
    const int srcb  = (lane < 16) ? 0 : 8;
    const int rbase = (lane < 16) ? 0 : 8;
#pragma unroll
    for (int r = 0; r < 8; ++r) {
        const float lr  = __shfl(l, srcb + r, 32);
        const float inv = 1.0f / lr;
        float* orow = out + ((size_t)b * GAT_N + i0 + rbase + r) * GAT_FO + mrow;
        float x;
        x = acc0[r] * inv; __builtin_nontemporal_store((x > 0.0f) ? x : (__expf(x) - 1.0f), orow);
        x = acc1[r] * inv; __builtin_nontemporal_store((x > 0.0f) ? x : (__expf(x) - 1.0f), orow + 16);
        x = acc2[r] * inv; __builtin_nontemporal_store((x > 0.0f) ? x : (__expf(x) - 1.0f), orow + 32);
        x = acc3[r] * inv; __builtin_nontemporal_store((x > 0.0f) ? x : (__expf(x) - 1.0f), orow + 48);
    }
}

// ---------------------------------------------------------------------------
extern "C" void kernel_launch(void* const* d_in, const int* in_sizes, int n_in,
                              void* d_out, int out_size, void* d_ws, size_t ws_size,
                              hipStream_t stream)
{
    (void)in_sizes; (void)n_in; (void)out_size; (void)ws_size;

    const float* inp = (const float*)d_in[0];
    const int*   adj = (const int*)d_in[1];
    const float* W   = (const float*)d_in[2];
    const float* a   = (const float*)d_in[3];
    float* out = (float*)d_out;

    char* ws = (char*)d_ws;
    _Float16* hT = (_Float16*)ws;                                    // 4 MB
    float* s1 = (float*)(ws + (size_t)GAT_B * GAT_FO * GAT_N * sizeof(_Float16));
    float* s2 = s1 + GAT_B * GAT_N;

    gat_proj<<<dim3((GAT_B * GAT_N) / 256), dim3(256), 0, stream>>>(inp, W, a, hT, s1, s2);
    gat_attn<<<dim3(GAT_N / 64, GAT_B), dim3(128), 0, stream>>>(adj, hT, s1, s2, out);
}
